// PositionAwareAttention_43155831390954
// MI455X (gfx1250) — matrix operand order repaired and checked by reading, not compiled, not run on hardware
//
#include <hip/hip_runtime.h>
#include <hip/hip_bf16.h>
#include <math.h>

// Problem sizes (fixed by the reference)
#define B_   32
#define S_   2048
#define DIN_ 1024
#define DQ_  512
#define DF_  64
#define DA_  256

#define ROWS_WG 128          // (b,s) rows per workgroup -> 8 waves x 16 rows
#define KSTEPS  34           // (1024 + 64) / 32
#define BSTRIDE 40           // LDS row stride in bf16 elems (80 B) == TDM pad layout
#define SLAB_ELTS (DA_ * 32) // 8192 bf16 per K-slab (16 KB)

typedef __attribute__((ext_vector_type(16))) __bf16 v16bf;
typedef __attribute__((ext_vector_type(8)))  float  v8f;
typedef __attribute__((ext_vector_type(4)))  unsigned int u32x4;
typedef __attribute__((ext_vector_type(8)))  int          i32x8;
typedef __attribute__((ext_vector_type(4)))  int          i32x4;

union BF16x16 {
    v16bf  b;
    __bf16 e[16];
    uint4  q[2];
};

// float -> bf16 RNE (used only in the one-time pack kernel)
__device__ __forceinline__ unsigned short f2bf(float f) {
    unsigned int u = __float_as_uint(f);
    unsigned int r = u + 0x7fffu + ((u >> 16) & 1u);
    return (unsigned short)(r >> 16);
}

// ---------------------------------------------------------------------------
// Kernel P: pack [Uw | Ww] -> bf16, K-slab-major: Wpack[ks][n][32]  (runs once
// per call; 544 KB). Removes ALL weight conversion work from the hot loop.
// ---------------------------------------------------------------------------
__global__ __launch_bounds__(DA_)
void paa_pack_kernel(const float* __restrict__ Uw, const float* __restrict__ Ww,
                     unsigned short* __restrict__ Wpack)
{
    const int ks = blockIdx.x;           // 0..33
    const int n  = threadIdx.x;          // 0..255
    const float* src = (ks < 32) ? (Uw + n * DIN_ + ks * 32)
                                 : (Ww + n * DF_  + (ks - 32) * 32);
    alignas(16) unsigned short tmp[32];
    #pragma unroll
    for (int v = 0; v < 8; ++v) {
        float4 d = reinterpret_cast<const float4*>(src)[v];
        tmp[v * 4 + 0] = f2bf(d.x);
        tmp[v * 4 + 1] = f2bf(d.y);
        tmp[v * 4 + 2] = f2bf(d.z);
        tmp[v * 4 + 3] = f2bf(d.w);
    }
    unsigned short* dst = Wpack + (size_t)ks * SLAB_ELTS + n * 32;
    #pragma unroll
    for (int v = 0; v < 4; ++v)
        reinterpret_cast<uint4*>(dst)[v] = reinterpret_cast<const uint4*>(tmp)[v];
}

// ---------------------------------------------------------------------------
// Kernel A: bias[b, a] = q[b,:]·Vw[a,:] + Vb[a] + Ub[a] + Wb[a]   (tiny)
// ---------------------------------------------------------------------------
__global__ __launch_bounds__(DA_)
void paa_bias_kernel(const float* __restrict__ q,  const float* __restrict__ Vw,
                     const float* __restrict__ Vb, const float* __restrict__ Ub,
                     const float* __restrict__ Wb, float* __restrict__ bias)
{
    const int b = blockIdx.x;
    const int a = threadIdx.x;
    const float* qr = q  + b * DQ_;
    const float* vr = Vw + a * DQ_;
    float s = 0.0f;
    for (int d = 0; d < DQ_; d += 4) {
        float4 qq = *reinterpret_cast<const float4*>(qr + d);
        float4 vv = *reinterpret_cast<const float4*>(vr + d);
        s += qq.x * vv.x + qq.y * vv.y + qq.z * vv.z + qq.w * vv.w;
    }
    bias[b * DA_ + a] = s + Vb[a] + Ub[a] + Wb[a];
}

// ---------------------------------------------------------------------------
// TDM: DMA one 16 KB bf16 weight slab into LDS.  Pad 4 dwords every 16 dwords
// -> LDS row stride 40 bf16 (80 B), bank-conflict-free for the 16-lane
// ds_load_b128 fragment reads.  (D# layout per CDNA5 ISA)
// ---------------------------------------------------------------------------
__device__ __forceinline__ void tdm_issue_slab(const unsigned short* gsrc,
                                               unsigned int lds_byte_off)
{
    const unsigned long long ga = (unsigned long long)(size_t)gsrc;
    u32x4 g0;
    g0[0] = 1u;                                       // count=1, is_restore=0
    g0[1] = lds_byte_off;                             // lds_addr
    g0[2] = (unsigned int)(ga & 0xffffffffu);         // global_addr[31:0]
    g0[3] = (unsigned int)((ga >> 32) & 0x1ffffffu)   // global_addr[56:32]
          | (2u << 30);                               // type=2 ("image")
    i32x8 g1;
    g1[0] = (int)((1u << 16)        // data_size = 2 bytes
                | (1u << 20)        // pad_enable
                | (3u << 22)        // pad_interval: 16 dwords of data
                | (3u << 25));      // pad_amount:   4 dwords of pad
    g1[1] = (int)((unsigned)SLAB_ELTS << 16);  // tensor_dim0[15:0] @ bits[63:48]
    g1[2] = (int)(1u << 16);                   // tensor_dim0 hi=0; tensor_dim1=1
    g1[3] = (int)((unsigned)SLAB_ELTS << 16);  // tile_dim0 = 8192 @ bits[127:112]
    g1[4] = 1;                                 // tile_dim1 = 1
    g1[5] = SLAB_ELTS;                         // tensor_dim0_stride (lo32)
    g1[6] = 0;
    g1[7] = 0;
    const i32x4 gz4 = {0, 0, 0, 0};
    const i32x8 gz8 = {0, 0, 0, 0, 0, 0, 0, 0};
    // clang-23 / therock-10.0 signature: (g0, g1, g2, g3, g4, cpol)
    __builtin_amdgcn_tensor_load_to_lds(g0, g1, gz4, gz4, gz8, 0);
}

// ---------------------------------------------------------------------------
// Kernel B: fused [x|f] @ [Uw|Ww]^T  (bf16 WMMA, f32 acc)  -> tanh -> ·Tw
//           -> masked scores[b, s]
// Grid: (S/128, B), block 256 (8 waves x 16 rows x 128 cols).
// Weight slabs arrive via double-buffered TDM (TENSORcnt), compute overlaps DMA.
// ---------------------------------------------------------------------------
__global__ __launch_bounds__(256)
void paa_scores_kernel(const float*          __restrict__ x,
                       const unsigned char*  __restrict__ x_mask,
                       const float*          __restrict__ f,
                       const unsigned short* __restrict__ Wpack,  // [34][256][32] bf16
                       const float*          __restrict__ Tw,
                       const float*          __restrict__ bias,   // [B, DA]
                       float*                __restrict__ scores) // [B, S]
{
    __shared__ unsigned short ldsB[2][DA_ * BSTRIDE];   // 2 x 20 KB slabs

    const int b    = blockIdx.y;
    const int s0   = blockIdx.x * ROWS_WG;
    const int tid  = threadIdx.x;
    const int lane = tid & 31;
    const int wave = tid >> 5;

    // A fragment (contiguous k order, same as the B fragment):
    //   lanes 0-15  : M = lane,    elems 0-7 -> K 0..7,  elems 8-15 -> K 8..15
    //   lanes 16-31 : M = lane-16, elems 0-7 -> K 16..23, elems 8-15 -> K 24..31
    const int mrow  = lane & 15;
    const int khalf = lane >> 4;
    const int srow  = s0 + wave * 16 + mrow;

    v8f acc[8];
    const v8f vzero = {0.f, 0.f, 0.f, 0.f, 0.f, 0.f, 0.f, 0.f};
    #pragma unroll
    for (int t = 0; t < 8; ++t) acc[t] = vzero;

    // Prime the pipeline: wave 0 launches the DMA of slab 0.
    if (tid < 32)
        tdm_issue_slab(Wpack, (unsigned int)(size_t)&ldsB[0][0]);

    for (int ks = 0; ks < KSTEPS; ++ks) {
        const int buf = ks & 1;

        if (tid < 32) {
            if (ks + 1 < KSTEPS) {     // prefetch next slab into the other buffer
                tdm_issue_slab(Wpack + (size_t)(ks + 1) * SLAB_ELTS,
                               (unsigned int)(size_t)&ldsB[buf ^ 1][0]);
                __builtin_amdgcn_s_wait_tensorcnt(1);   // slab ks landed
            } else {
                __builtin_amdgcn_s_wait_tensorcnt(0);
            }
        }
        __syncthreads();               // publish slab ks to all 8 waves

        // ---- per-lane A fragment from global (fp32 -> bf16 hardware cvt)
        const int  k0  = ks * 32;
        const bool inX = (k0 < DIN_);
        BF16x16 afrag;
        {
            const float* arow = inX
                ? (x + (size_t)(b * S_ + srow) * DIN_ + k0)
                : (f + (size_t)(b * S_ + srow) * DF_  + (k0 - DIN_));
            __builtin_prefetch(arow + 32, 0, 3);        // next K-slab of this row
            const int ka0 = khalf * 16;
            const int ka1 = khalf * 16 + 8;
            float4 d0 = reinterpret_cast<const float4*>(arow + ka0)[0];
            float4 d1 = reinterpret_cast<const float4*>(arow + ka0)[1];
            float4 d2 = reinterpret_cast<const float4*>(arow + ka1)[0];
            float4 d3 = reinterpret_cast<const float4*>(arow + ka1)[1];
            afrag.e[0]  = (__bf16)d0.x; afrag.e[1]  = (__bf16)d0.y;
            afrag.e[2]  = (__bf16)d0.z; afrag.e[3]  = (__bf16)d0.w;
            afrag.e[4]  = (__bf16)d1.x; afrag.e[5]  = (__bf16)d1.y;
            afrag.e[6]  = (__bf16)d1.z; afrag.e[7]  = (__bf16)d1.w;
            afrag.e[8]  = (__bf16)d2.x; afrag.e[9]  = (__bf16)d2.y;
            afrag.e[10] = (__bf16)d2.z; afrag.e[11] = (__bf16)d2.w;
            afrag.e[12] = (__bf16)d3.x; afrag.e[13] = (__bf16)d3.y;
            afrag.e[14] = (__bf16)d3.z; afrag.e[15] = (__bf16)d3.w;
        }

        // ---- 8 WMMAs across N columns 0..127 (of 256)
        const int kb = khalf * 16;   // B frag: lanes 0-15 -> K 0..15, 16-31 -> 16..31
        const int nb = lane & 15;
        #pragma unroll
        for (int t = 0; t < 8; ++t) {
            BF16x16 bfrag;
            const unsigned short* bp = &ldsB[buf][(t * 16 + nb) * BSTRIDE + kb];
            bfrag.q[0] = *reinterpret_cast<const uint4*>(bp);
            bfrag.q[1] = *reinterpret_cast<const uint4*>(bp + 8);
            acc[t] = __builtin_amdgcn_wmma_f32_16x16x32_bf16(
                false, afrag.b, false, bfrag.b, (short)0, acc[t], false, false);
        }
        __syncthreads();   // all reads of ldsB[buf] done before TDM re-fills it
    }

    // ---- epilogue: score[m] = sum_n tanh(acc + bias[n]) * Tw[n]
    // C layout: VGPR r, lanes 0-15 -> (M=r, N=lane); lanes 16-31 -> (M=r+8, N=lane-16)
    float rowsum[8];
    #pragma unroll
    for (int r = 0; r < 8; ++r) rowsum[r] = 0.0f;
    #pragma unroll
    for (int t = 0; t < 8; ++t) {
        const int   n   = t * 16 + (lane & 15);
        const float bia = bias[b * DA_ + n];
        const float tw  = Tw[n];
        #pragma unroll
        for (int r = 0; r < 8; ++r)
            rowsum[r] += tanhf(acc[t][r] + bia) * tw;
    }
    #pragma unroll
    for (int r = 0; r < 8; ++r) {
        rowsum[r] += __shfl_xor(rowsum[r], 1, 16);
        rowsum[r] += __shfl_xor(rowsum[r], 2, 16);
        rowsum[r] += __shfl_xor(rowsum[r], 4, 16);
        rowsum[r] += __shfl_xor(rowsum[r], 8, 16);
    }
    if ((lane & 15) == 0) {
        #pragma unroll
        for (int r = 0; r < 8; ++r) {
            const int m = r + khalf * 8;
            const int s = s0 + wave * 16 + m;
            scores[b * S_ + s] = x_mask[b * S_ + s] ? -INFINITY : rowsum[r];
        }
    }
}

// ---------------------------------------------------------------------------
// Kernel C: softmax over S per batch row
// ---------------------------------------------------------------------------
__global__ __launch_bounds__(256)
void paa_softmax_kernel(const float* __restrict__ scores, float* __restrict__ weights)
{
    __shared__ float red[8];
    const int b   = blockIdx.x;
    const int tid = threadIdx.x;
    const float* src = scores + b * S_;

    float v[8];
    float mx = -INFINITY;
    #pragma unroll
    for (int i = 0; i < 8; ++i) {
        v[i] = src[tid * 8 + i];
        mx = fmaxf(mx, v[i]);
    }
    for (int off = 1; off < 32; off <<= 1) mx = fmaxf(mx, __shfl_xor(mx, off, 32));
    if ((tid & 31) == 0) red[tid >> 5] = mx;
    __syncthreads();
    float gmx = red[0];
    #pragma unroll
    for (int w = 1; w < 8; ++w) gmx = fmaxf(gmx, red[w]);
    __syncthreads();

    float sum = 0.0f;
    #pragma unroll
    for (int i = 0; i < 8; ++i) { v[i] = __expf(v[i] - gmx); sum += v[i]; }
    for (int off = 1; off < 32; off <<= 1) sum += __shfl_xor(sum, off, 32);
    if ((tid & 31) == 0) red[tid >> 5] = sum;
    __syncthreads();
    float gs = 0.0f;
    #pragma unroll
    for (int w = 0; w < 8; ++w) gs += red[w];

    const float inv = 1.0f / gs;
    #pragma unroll
    for (int i = 0; i < 8; ++i) weights[b * S_ + tid * 8 + i] = v[i] * inv;
}

// ---------------------------------------------------------------------------
// Kernel D: out[b, d] = sum_s weights[b,s] * x[b,s,d]   (streams x once)
// ---------------------------------------------------------------------------
__global__ __launch_bounds__(128)
void paa_output_kernel(const float* __restrict__ x,
                       const float* __restrict__ weights,
                       float*       __restrict__ out)
{
    const int b = blockIdx.y;
    const int d = blockIdx.x * 128 + threadIdx.x;
    const float* xb = x + (size_t)b * S_ * DIN_ + d;
    const float* wb = weights + b * S_;
    float a0 = 0.f, a1 = 0.f, a2 = 0.f, a3 = 0.f;
    for (int s = 0; s < S_; s += 4) {
        a0 += wb[s + 0] * xb[(size_t)(s + 0) * DIN_];
        a1 += wb[s + 1] * xb[(size_t)(s + 1) * DIN_];
        a2 += wb[s + 2] * xb[(size_t)(s + 2) * DIN_];
        a3 += wb[s + 3] * xb[(size_t)(s + 3) * DIN_];
    }
    out[b * DIN_ + d] = (a0 + a1) + (a2 + a3);
}

// ---------------------------------------------------------------------------
extern "C" void kernel_launch(void* const* d_in, const int* in_sizes, int n_in,
                              void* d_out, int out_size, void* d_ws, size_t ws_size,
                              hipStream_t stream)
{
    const float*         x      = (const float*)d_in[0];
    const unsigned char* x_mask = (const unsigned char*)d_in[1];  // jax bool: 1 byte
    const float*         f      = (const float*)d_in[2];
    const float*         q      = (const float*)d_in[3];
    const float*         Uw     = (const float*)d_in[4];
    const float*         Ub     = (const float*)d_in[5];
    const float*         Vw     = (const float*)d_in[6];
    const float*         Vb     = (const float*)d_in[7];
    const float*         Ww     = (const float*)d_in[8];
    const float*         Wb     = (const float*)d_in[9];
    const float*         Tw     = (const float*)d_in[10];

    float* out_vec = (float*)d_out;            // [B, DIN]
    float* weights = out_vec + B_ * DIN_;      // [B, S] (second tuple output)

    float*          bias_ws = (float*)d_ws;                    // [B, DA]
    float*          scores  = bias_ws + B_ * DA_;              // [B, S]
    unsigned short* Wpack   = (unsigned short*)(scores + B_ * S_); // [34][256][32] bf16

    paa_pack_kernel<<<dim3(KSTEPS), dim3(DA_), 0, stream>>>(Uw, Ww, Wpack);
    paa_bias_kernel<<<dim3(B_), dim3(DA_), 0, stream>>>(q, Vw, Vb, Ub, Wb, bias_ws);
    paa_scores_kernel<<<dim3(S_ / ROWS_WG, B_), dim3(256), 0, stream>>>(
        x, x_mask, f, Wpack, Tw, bias_ws, scores);
    paa_softmax_kernel<<<dim3(B_), dim3(256), 0, stream>>>(scores, weights);
    paa_output_kernel<<<dim3(DIN_ / 128, B_), dim3(128), 0, stream>>>(x, weights, out_vec);
}
